// pwc_residual_84172769067803
// MI455X (gfx1250) — matrix operand order repaired and checked by reading, not compiled, not run on hardware
//
#include <hip/hip_runtime.h>
#include <hip/hip_bf16.h>

// ---------------------------------------------------------------------------
// MI455X (gfx1250) implementation of the ada_mask PWC-residual reference.
// All 3x3 convs run as implicit GEMM on v_wmma_f32_16x16x32_bf16 (wave32).
// Weight (B) fragments are staged to LDS via the Tensor Data Mover
// (tensor_load_to_lds + s_wait_tensorcnt) and shared by the 4 waves/block.
// Zero-padding of A uses a zero page + address select (no EXEC branching).
// ---------------------------------------------------------------------------

typedef __attribute__((ext_vector_type(16))) __bf16 v16bf;
typedef __attribute__((ext_vector_type(8)))  __bf16 v8bf;
typedef __attribute__((ext_vector_type(8)))  float  v8f;
typedef __attribute__((ext_vector_type(4)))  unsigned int u32x4;
typedef __attribute__((ext_vector_type(8)))  unsigned int u32x8;

#define BN 4
#define HH 256
#define WW 256

// ---------------------------------------------------------------------------
// 0) Zero a small region (zero page for padded conv taps)
// ---------------------------------------------------------------------------
__global__ __launch_bounds__(256) void k_zero(__hip_bfloat16* __restrict__ p, int n) {
  int i = blockIdx.x * 256 + threadIdx.x;
  if (i < n) p[i] = __float2bfloat16(0.f);
}

// ---------------------------------------------------------------------------
// 1) Flow upsample x4 (bilinear, half-pixel centers, edge clamp) * 20
// ---------------------------------------------------------------------------
__global__ __launch_bounds__(256) void k_flow_up(const float* __restrict__ fl,
                                                 float* __restrict__ flow,
                                                 int Hl, int Wl) {
  int i = blockIdx.x * 256 + threadIdx.x;
  int total = BN * 2 * HH * WW;
  if (i >= total) return;
  int x = i % WW, y = (i / WW) % HH, c = (i / (WW * HH)) % 2, b = i / (WW * HH * 2);
  float sx = (x + 0.5f) * 0.25f - 0.5f;
  float sy = (y + 0.5f) * 0.25f - 0.5f;
  float x0f = floorf(sx), y0f = floorf(sy);
  int x0 = (int)x0f, y0 = (int)y0f;
  float wx = sx - x0f, wy = sy - y0f;
  int x0c = min(max(x0, 0), Wl - 1), x1c = min(max(x0 + 1, 0), Wl - 1);
  int y0c = min(max(y0, 0), Hl - 1), y1c = min(max(y0 + 1, 0), Hl - 1);
  const float* p = fl + ((size_t)(b * 2 + c) * Hl) * Wl;
  float v = p[y0c * Wl + x0c] * (1 - wx) * (1 - wy) + p[y0c * Wl + x1c] * wx * (1 - wy) +
            p[y1c * Wl + x0c] * (1 - wx) * wy + p[y1c * Wl + x1c] * wx * wy;
  flow[i] = v * 20.0f;
}

// ---------------------------------------------------------------------------
// 2) Warp ref (zeros padding, thresholded ones-mask), build 11ch features
//    into NHWC bf16 padded to 32 channels. Writes warp_ref (f32 NCHW) too.
// ---------------------------------------------------------------------------
__global__ __launch_bounds__(256) void k_warp_feat(const float* __restrict__ ref,
                                                   const float* __restrict__ sr,
                                                   const float* __restrict__ flow,
                                                   float* __restrict__ warp_out,
                                                   float* __restrict__ maskref,
                                                   __hip_bfloat16* __restrict__ feat) {
  int i = blockIdx.x * 256 + threadIdx.x;
  int total = BN * HH * WW;
  if (i >= total) return;
  int x = i % WW, y = (i / WW) % HH, b = i / (HH * WW);
  size_t plane = (size_t)HH * WW;
  float flx = flow[((size_t)b * 2 + 0) * plane + (size_t)y * WW + x];
  float fly = flow[((size_t)b * 2 + 1) * plane + (size_t)y * WW + x];
  float fx = x + flx, fy = y + fly;
  float x0f = floorf(fx), y0f = floorf(fy);
  int x0 = (int)x0f, y0 = (int)y0f;
  float wx = fx - x0f, wy = fy - y0f;
  float w00 = (1 - wx) * (1 - wy), w10 = wx * (1 - wy);
  float w01 = (1 - wx) * wy, w11 = wx * wy;
  bool vx0 = (x0 >= 0) & (x0 < WW), vx1 = (x0 + 1 >= 0) & (x0 + 1 < WW);
  bool vy0 = (y0 >= 0) & (y0 < HH), vy1 = (y0 + 1 >= 0) & (y0 + 1 < HH);
  float m = (vx0 && vy0 ? w00 : 0.f) + (vx1 && vy0 ? w10 : 0.f) +
            (vx0 && vy1 ? w01 : 0.f) + (vx1 && vy1 ? w11 : 0.f);
  float mth = (m >= 0.9999f) ? 1.f : 0.f;
  maskref[i] = mth;
  int x0c = min(max(x0, 0), WW - 1), x1c = min(max(x0 + 1, 0), WW - 1);
  int y0c = min(max(y0, 0), HH - 1), y1c = min(max(y0 + 1, 0), HH - 1);
  __hip_bfloat16* f = feat + (size_t)i * 32;
  for (int c = 0; c < 3; ++c) {
    const float* p = ref + ((size_t)b * 3 + c) * plane;
    float s = (vx0 && vy0 ? p[y0c * WW + x0c] * w00 : 0.f) +
              (vx1 && vy0 ? p[y0c * WW + x1c] * w10 : 0.f) +
              (vx0 && vy1 ? p[y1c * WW + x0c] * w01 : 0.f) +
              (vx1 && vy1 ? p[y1c * WW + x1c] * w11 : 0.f);
    s *= mth;
    warp_out[((size_t)b * 3 + c) * plane + (size_t)y * WW + x] = s;
    float sv = sr[((size_t)b * 3 + c) * plane + (size_t)y * WW + x];
    f[c] = __float2bfloat16(s);
    f[3 + c] = __float2bfloat16(sv);
    f[8 + c] = __float2bfloat16(s - sv);
  }
  f[6] = __float2bfloat16(flx);
  f[7] = __float2bfloat16(fly);
  for (int c = 11; c < 32; ++c) f[c] = __float2bfloat16(0.f);
}

// ---------------------------------------------------------------------------
// 3) Weight repack: fp32 OIHW -> bf16 WMMA B-fragment order.
//    Layout: [coutTile][tap][cinChunk][lane 0..31][half 0..15]
//    ISA B layout (16-bit 32x16): lane L -> N = L%16, half h -> K = (L/16)*8+(h&7)+(h>>3)*16
// ---------------------------------------------------------------------------
__global__ __launch_bounds__(256) void k_pack(const float* __restrict__ w,
                                              __hip_bfloat16* __restrict__ wpk,
                                              int cinReal, int cout,
                                              int cinChunks, int coutTiles) {
  size_t i = (size_t)blockIdx.x * 256 + threadIdx.x;
  size_t total = (size_t)coutTiles * 9 * cinChunks * 512;
  if (i >= total) return;
  int h = (int)(i & 15);
  int lane = (int)((i >> 4) & 31);
  size_t r = i >> 9;
  int cc = (int)(r % cinChunks); r /= cinChunks;
  int tap = (int)(r % 9); r /= 9;
  int nt = (int)r;
  int N = nt * 16 + (lane & 15);
  int cin = cc * 32 + (lane >> 4) * 8 + (h & 7) + (h >> 3) * 16;
  float v = 0.f;
  if (N < cout && cin < cinReal) v = w[((size_t)N * cinReal + cin) * 9 + tap];
  wpk[i] = __float2bfloat16(v);
}

// ---------------------------------------------------------------------------
// 4) Conv3x3 implicit GEMM on v_wmma_f32_16x16x32_bf16.
//    One wave -> 16 pixels x 16 cout; 4 waves/block share one cout tile.
//    Per tap, the B-fragment block (cinChunks KB) is DMA'd to LDS by the
//    Tensor Data Mover and read back with ds_load_b128 by all 4 waves.
//    Padded taps read a zeroed page via per-lane address select (no branch).
// ---------------------------------------------------------------------------
__global__ __launch_bounds__(128) void k_conv3x3(
    const __hip_bfloat16* __restrict__ in, const __hip_bfloat16* __restrict__ wpk,
    const float* __restrict__ bias,
    __hip_bfloat16* __restrict__ out, float* __restrict__ out32,
    const __hip_bfloat16* __restrict__ res,
    const __hip_bfloat16* __restrict__ zpage,
    int Hin, int Win, int CinPad, int cinChunks,
    int Hout, int Wout, int CoutPad, int CoutReal,
    int stride, int relu) {
  __shared__ __hip_bfloat16 smem[8192];  // 16KB: one tap's B fragments (<=16 chunks)
  int lane = threadIdx.x & 31;
  int wid = threadIdx.x >> 5;
  int tile = blockIdx.x * 4 + wid;
  int nt = blockIdx.y;
  int b = blockIdx.z;
  int g = lane >> 4, mrow = lane & 15;
  int p = tile * 16 + mrow;
  int oy = p / Wout, ox = p % Wout;
  int iy0 = oy * stride - 1, ix0 = ox * stride - 1;
  const __hip_bfloat16* abase = in + ((size_t)b * Hin) * Win * CinPad + (size_t)g * 8;
  const __hip_bfloat16* wbase = wpk + (size_t)nt * 9 * cinChunks * 512;
  unsigned int ldsBase = (unsigned int)(size_t)(&smem[0]);  // LDS byte offset
  v8f acc = {};
  for (int tap = 0; tap < 9; ++tap) {
    if (wid == 0) {
      // Tensor DMA descriptor (D#), 2-group form (<=2D tensor):
      // 1-D copy of cinChunks*1024 bytes as data_size=8B elements.
      unsigned long long ga =
          (unsigned long long)(size_t)(wbase + (size_t)tap * cinChunks * 512);
      unsigned int Nu = (unsigned int)cinChunks * 128u;  // 8-byte units
      u32x4 g0;
      u32x8 g1;
      g0[0] = 1u;                      // count=1, user descriptor
      g0[1] = ldsBase;                 // lds_addr
      g0[2] = (unsigned int)ga;        // global_addr[31:0]
      g0[3] = (unsigned int)((ga >> 32) & 0x1FFFFFFu) | (2u << 30);  // addr hi | type=2
      g1[0] = (3u << 16);              // wg_mask=0, data_size=8B
      g1[1] = (Nu & 0xFFFFu) << 16;    // atomic_addr=0, tensor_dim0 lo16
      g1[2] = (1u << 16);              // tensor_dim0 hi16=0, tensor_dim1=1 (lo16)
      g1[3] = (Nu & 0xFFFFu) << 16;    // tensor_dim1 hi16=0, tile_dim0=Nu
      g1[4] = 0u;                      // tile_dim1=0 (unused), tile_dim2=0
      g1[5] = Nu;                      // tensor_dim0_stride lo32
      g1[6] = 0u;                      // stride hi16, dim1_stride lo16
      g1[7] = 0u;                      // dim1_stride hi32
      asm volatile("tensor_load_to_lds %0, %1" ::"s"(g0), "s"(g1) : "memory");
      __builtin_amdgcn_s_wait_tensorcnt(0);
    }
    __syncthreads();
    int iy = iy0 + tap / 3;
    int ix = ix0 + tap % 3;
    bool iv = (iy >= 0) & (iy < Hin) & (ix >= 0) & (ix < Win);
    // Per-lane address select: OOB taps read zeros from the zero page.
    const __hip_bfloat16* arow =
        iv ? (abase + ((size_t)iy * Win + ix) * CinPad) : zpage;
    for (int cc = 0; cc < cinChunks; ++cc) {
      v8bf lo = *(const v8bf*)(arow + cc * 32);       // K = g*8 + 0..7
      v8bf hi = *(const v8bf*)(arow + cc * 32 + 16);  // K = g*8 + 16..23
      v16bf a = __builtin_shufflevector(lo, hi, 0, 1, 2, 3, 4, 5, 6, 7,
                                        8, 9, 10, 11, 12, 13, 14, 15);
      v16bf wb = *(const v16bf*)(smem + (size_t)cc * 512 + lane * 16);  // ds_load_b128 x2
      acc = __builtin_amdgcn_wmma_f32_16x16x32_bf16(false, a, false, wb,
                                                    (short)0, acc, false, false);
    }
    __syncthreads();
  }
  int co = nt * 16 + mrow;  // C/D layout: N = lane%16 (same for all 8 accs)
  float bb = (co < CoutReal) ? bias[co] : 0.f;
#pragma unroll
  for (int v = 0; v < 8; ++v) {
    int M = v + 8 * g;  // C/D layout: row M = vgpr + 8*(lane/16)
    int p2 = tile * 16 + M;
    int oy2 = p2 / Wout, ox2 = p2 % Wout;
    float val = acc[v] + bb;
    if (relu) val = fmaxf(val, 0.f);
    size_t pix = ((size_t)b * Hout + oy2) * Wout + ox2;
    if (res) val += __bfloat162float(res[pix * CoutPad + co]);
    if (out32) {
      if (co < CoutReal)
        out32[((size_t)b * CoutReal + co) * Hout * Wout + (size_t)oy2 * Wout + ox2] = val;
    } else {
      out[pix * CoutPad + co] = __float2bfloat16(val);
    }
  }
}

// ---------------------------------------------------------------------------
// 5) 2x bilinear upsample, align_corners=True, NHWC bf16
// ---------------------------------------------------------------------------
__global__ __launch_bounds__(256) void k_up2(const __hip_bfloat16* __restrict__ in,
                                             __hip_bfloat16* __restrict__ out,
                                             int hin, int win, int C) {
  size_t i = (size_t)blockIdx.x * 256 + threadIdx.x;
  int ho = 2 * hin, wo = 2 * win;
  size_t total = (size_t)BN * ho * wo * C;
  if (i >= total) return;
  int c = (int)(i % C);
  size_t r = i / C;
  int x = (int)(r % wo); r /= wo;
  int y = (int)(r % ho);
  int b = (int)(r / ho);
  float sy = y * (float)(hin - 1) / (float)(ho - 1);
  float sx = x * (float)(win - 1) / (float)(wo - 1);
  int y0 = (int)floorf(sy), x0 = (int)floorf(sx);
  int y1 = min(y0 + 1, hin - 1), x1 = min(x0 + 1, win - 1);
  float wy = sy - y0, wx = sx - x0;
  const __hip_bfloat16* p = in + (size_t)b * hin * win * C;
  float v00 = __bfloat162float(p[((size_t)y0 * win + x0) * C + c]);
  float v10 = __bfloat162float(p[((size_t)y0 * win + x1) * C + c]);
  float v01 = __bfloat162float(p[((size_t)y1 * win + x0) * C + c]);
  float v11 = __bfloat162float(p[((size_t)y1 * win + x1) * C + c]);
  float v = v00 * (1 - wy) * (1 - wx) + v10 * (1 - wy) * wx +
            v01 * wy * (1 - wx) + v11 * wy * wx;
  out[i] = __float2bfloat16(v);
}

// ---------------------------------------------------------------------------
// 6) Final blend: 25-patch weighted sum of replication-padded warp_ref
// ---------------------------------------------------------------------------
__global__ __launch_bounds__(256) void k_final(const float* __restrict__ sr,
                                               const float* __restrict__ warp,
                                               const float* __restrict__ mask,
                                               const float* __restrict__ maskref,
                                               float* __restrict__ out) {
  int i = blockIdx.x * 256 + threadIdx.x;
  int total = BN * HH * WW;
  if (i >= total) return;
  int x = i % WW, y = (i / WW) % HH, b = i / (HH * WW);
  size_t plane = (size_t)HH * WW;
  const float* mk = mask + (size_t)b * 26 * plane + (size_t)y * WW + x;
  float msig = (1.f / (1.f + expf(-mk[25 * plane]))) * maskref[i];
  float acc0 = 0.f, acc1 = 0.f, acc2 = 0.f;
  for (int t = 0; t < 25; ++t) {
    int dy = t / 5 - 2, dx = t % 5 - 2;
    int yy = min(max(y + dy, 0), HH - 1);
    int xx = min(max(x + dx, 0), WW - 1);
    float mw = mk[(size_t)t * plane];
    size_t q = (size_t)yy * WW + xx;
    acc0 += warp[((size_t)b * 3 + 0) * plane + q] * mw;
    acc1 += warp[((size_t)b * 3 + 1) * plane + q] * mw;
    acc2 += warp[((size_t)b * 3 + 2) * plane + q] * mw;
  }
  float accs[3] = {acc0, acc1, acc2};
  for (int c = 0; c < 3; ++c) {
    float s = sr[((size_t)b * 3 + c) * plane + (size_t)y * WW + x];
    out[((size_t)b * 3 + c) * plane + (size_t)y * WW + x] =
        accs[c] * msig + s * (1.f - msig);
  }
}

// ---------------------------------------------------------------------------
// Host orchestration.
// Param flattening assumption: JAX pytree order = dict keys sorted
// alphabetically, leaves within a layer dict also sorted (b,w / b1,b2,w1,w2):
//   0 ref, 1 sr, 2 flow_lr,
//   3 d1.b 4 d1.w | 5 d2.b 6 d2.w | 7 d3.b 8 d3.w | 9 head.b 10 head.w
//   11-14 r1{b1,b2,w1,w2} | 15-18 r1_1d | 19-22 r1_1u | 23-26 r1_2d | 27-30 r1_2u
//   31-34 r2 | 35-38 r2_1d | 39-42 r2_1u | 43-46 r2_2d | 47-50 r2_2u
//   51-54 r3_1d | 55-58 r3_1u | 59-62 r3_2d | 63-66 r3_2u | 67 tail.b 68 tail.w
// ---------------------------------------------------------------------------
extern "C" void kernel_launch(void* const* d_in, const int* in_sizes, int n_in,
                              void* d_out, int out_size, void* d_ws, size_t ws_size,
                              hipStream_t stream) {
  (void)in_sizes; (void)n_in; (void)out_size; (void)ws_size;
  const float* ref = (const float*)d_in[0];
  const float* sr = (const float*)d_in[1];
  const float* flow_lr = (const float*)d_in[2];

  float* out_final = (float*)d_out;
  float* warp_out = out_final + (size_t)BN * 3 * HH * WW;      // 786432
  float* mask_out = out_final + (size_t)2 * BN * 3 * HH * WW;  // NCHW [B,26,H,W]

  size_t off = 0;
  auto alloc = [&](size_t bytes) -> void* {
    void* p = (char*)d_ws + off;
    off += (bytes + 255) & ~(size_t)255;
    return p;
  };

  __hip_bfloat16* zpage = (__hip_bfloat16*)alloc(4096);  // zero page for padded taps
  float* flow = (float*)alloc((size_t)BN * 2 * HH * WW * 4);
  float* maskref = (float*)alloc((size_t)BN * HH * WW * 4);
  __hip_bfloat16* feat = (__hip_bfloat16*)alloc((size_t)BN * HH * WW * 32 * 2);
  __hip_bfloat16* b64a = (__hip_bfloat16*)alloc((size_t)BN * 256 * 256 * 64 * 2);
  __hip_bfloat16* b64b = (__hip_bfloat16*)alloc((size_t)BN * 256 * 256 * 64 * 2);
  __hip_bfloat16* b64c = (__hip_bfloat16*)alloc((size_t)BN * 256 * 256 * 64 * 2);
  __hip_bfloat16* b128a = (__hip_bfloat16*)alloc((size_t)BN * 128 * 128 * 128 * 2);
  __hip_bfloat16* b128b = (__hip_bfloat16*)alloc((size_t)BN * 128 * 128 * 128 * 2);
  __hip_bfloat16* b128c = (__hip_bfloat16*)alloc((size_t)BN * 128 * 128 * 128 * 2);
  __hip_bfloat16* b256a = (__hip_bfloat16*)alloc((size_t)BN * 64 * 64 * 256 * 2);
  __hip_bfloat16* b256b = (__hip_bfloat16*)alloc((size_t)BN * 64 * 64 * 256 * 2);
  __hip_bfloat16* b256c = (__hip_bfloat16*)alloc((size_t)BN * 64 * 64 * 256 * 2);
  __hip_bfloat16* b512a = (__hip_bfloat16*)alloc((size_t)BN * 32 * 32 * 512 * 2);
  __hip_bfloat16* b512b = (__hip_bfloat16*)alloc((size_t)BN * 32 * 32 * 512 * 2);
  __hip_bfloat16* b512c = (__hip_bfloat16*)alloc((size_t)BN * 32 * 32 * 512 * 2);
  __hip_bfloat16* up512 = (__hip_bfloat16*)alloc((size_t)BN * 64 * 64 * 512 * 2);
  __hip_bfloat16* up256 = (__hip_bfloat16*)alloc((size_t)BN * 128 * 128 * 256 * 2);
  __hip_bfloat16* up128 = (__hip_bfloat16*)alloc((size_t)BN * 256 * 256 * 128 * 2);

  // front-end
  k_zero<<<8, 256, 0, stream>>>(zpage, 2048);
  {
    int total = BN * 2 * HH * WW;
    k_flow_up<<<(total + 255) / 256, 256, 0, stream>>>(flow_lr, flow, HH / 4, WW / 4);
  }
  {
    int total = BN * HH * WW;
    k_warp_feat<<<(total + 255) / 256, 256, 0, stream>>>(ref, sr, flow, warp_out,
                                                         maskref, feat);
  }

  // conv helper: pack weights then run WMMA conv
  auto convL = [&](int widx, int bidx, int cin, int cout, int Hin, int Win,
                   int stride, int relu, const void* inp, void* outp,
                   const void* resp, float* out32p) {
    int cinChunks = (cin + 31) / 32;
    int coutTiles = (cout + 15) / 16;
    size_t pkElems = (size_t)coutTiles * 9 * cinChunks * 512;
    __hip_bfloat16* wpk = (__hip_bfloat16*)alloc(pkElems * 2);
    k_pack<<<(unsigned)((pkElems + 255) / 256), 256, 0, stream>>>(
        (const float*)d_in[widx], wpk, cin, cout, cinChunks, coutTiles);
    int Hout = (stride == 2) ? Hin / 2 : Hin;
    int Wout = (stride == 2) ? Win / 2 : Win;
    dim3 grid(Hout * Wout / 64, coutTiles, BN);
    k_conv3x3<<<grid, 128, 0, stream>>>(
        (const __hip_bfloat16*)inp, wpk, (const float*)d_in[bidx],
        (__hip_bfloat16*)outp, out32p, (const __hip_bfloat16*)resp, zpage,
        Hin, Win, cinChunks * 32, cinChunks, Hout, Wout, cout, cout, stride, relu);
  };
  auto up2 = [&](const __hip_bfloat16* inb, __hip_bfloat16* outb, int hin, int win, int C) {
    size_t total = (size_t)BN * 2 * hin * 2 * win * C;
    k_up2<<<(unsigned)((total + 255) / 256), 256, 0, stream>>>(inb, outb, hin, win, C);
  };

  // ---- encoder ----
  convL(10, 9, 11, 64, 256, 256, 1, 0, feat, b64b, nullptr, nullptr);      // head
  convL(13, 11, 64, 64, 256, 256, 1, 1, b64b, b64c, nullptr, nullptr);     // r1.c1
  convL(14, 12, 64, 64, 256, 256, 1, 0, b64c, b64b, nullptr, nullptr);     // r1.c2
  convL(33, 31, 64, 64, 256, 256, 1, 1, b64b, b64c, nullptr, nullptr);     // r2.c1
  convL(34, 32, 64, 64, 256, 256, 1, 0, b64c, b64a, nullptr, nullptr);     // r2.c2 -> d0
  convL(4, 3, 64, 128, 256, 256, 2, 0, b64a, b128b, nullptr, nullptr);     // d1 (s2)
  convL(17, 15, 128, 128, 128, 128, 1, 1, b128b, b128c, nullptr, nullptr); // r1_1d.c1
  convL(18, 16, 128, 128, 128, 128, 1, 0, b128c, b128b, nullptr, nullptr); // r1_1d.c2
  convL(25, 23, 128, 128, 128, 128, 1, 1, b128b, b128c, nullptr, nullptr); // r1_2d.c1
  convL(26, 24, 128, 128, 128, 128, 1, 0, b128c, b128a, nullptr, nullptr); // -> d1
  convL(6, 5, 128, 256, 128, 128, 2, 0, b128a, b256b, nullptr, nullptr);   // d2 (s2)
  convL(37, 35, 256, 256, 64, 64, 1, 1, b256b, b256c, nullptr, nullptr);   // r2_1d.c1
  convL(38, 36, 256, 256, 64, 64, 1, 0, b256c, b256b, nullptr, nullptr);   // r2_1d.c2
  convL(45, 43, 256, 256, 64, 64, 1, 1, b256b, b256c, nullptr, nullptr);   // r2_2d.c1
  convL(46, 44, 256, 256, 64, 64, 1, 0, b256c, b256a, nullptr, nullptr);   // -> d2
  convL(8, 7, 256, 512, 64, 64, 2, 0, b256a, b512b, nullptr, nullptr);     // d3 (s2)
  convL(53, 51, 512, 512, 32, 32, 1, 1, b512b, b512c, nullptr, nullptr);   // r3_1d.c1
  convL(54, 52, 512, 512, 32, 32, 1, 0, b512c, b512b, nullptr, nullptr);   // r3_1d.c2
  convL(61, 59, 512, 512, 32, 32, 1, 1, b512b, b512c, nullptr, nullptr);   // r3_2d.c1
  convL(62, 60, 512, 512, 32, 32, 1, 0, b512c, b512a, nullptr, nullptr);   // -> d3

  // ---- decoder ----
  up2(b512a, up512, 32, 32, 512);                                          // up(d3)
  convL(57, 55, 512, 256, 64, 64, 1, 1, up512, b256b, nullptr, nullptr);   // r3_1u.c1
  convL(58, 56, 256, 256, 64, 64, 1, 0, b256b, b256c, nullptr, nullptr);   // r3_1u.c2
  convL(65, 63, 256, 256, 64, 64, 1, 1, b256c, b256b, nullptr, nullptr);   // r3_2u.c1
  convL(66, 64, 256, 256, 64, 64, 1, 0, b256b, b256c, b256a, nullptr);     // +d2 -> u2
  up2(b256c, up256, 64, 64, 256);                                          // up(u2)
  convL(41, 39, 256, 128, 128, 128, 1, 1, up256, b128b, nullptr, nullptr); // r2_1u.c1
  convL(42, 40, 128, 128, 128, 128, 1, 0, b128b, b128c, nullptr, nullptr); // r2_1u.c2
  convL(49, 47, 128, 128, 128, 128, 1, 1, b128c, b128b, nullptr, nullptr); // r2_2u.c1
  convL(50, 48, 128, 128, 128, 128, 1, 0, b128b, b128c, b128a, nullptr);   // +d1 -> u1
  up2(b128c, up128, 128, 128, 128);                                        // up(u1)
  convL(21, 19, 128, 64, 256, 256, 1, 1, up128, b64b, nullptr, nullptr);   // r1_1u.c1
  convL(22, 20, 64, 64, 256, 256, 1, 0, b64b, b64c, nullptr, nullptr);     // r1_1u.c2
  convL(29, 27, 64, 64, 256, 256, 1, 1, b64c, b64b, nullptr, nullptr);     // r1_2u.c1
  convL(30, 28, 64, 64, 256, 256, 1, 0, b64b, b64c, b64a, nullptr);        // +d0 -> u0
  convL(68, 67, 64, 26, 256, 256, 1, 0, b64c, nullptr, nullptr, mask_out); // tail (f32 NCHW)

  // ---- blend ----
  {
    int total = BN * HH * WW;
    k_final<<<(total + 255) / 256, 256, 0, stream>>>(sr, warp_out, mask_out,
                                                     maskref, out_final);
  }
}
